// QuantizedLinear_43353399885902
// MI455X (gfx1250) — 8-bit matrix operand order repaired and checked by reading, compiled, not run on hardware
//
#include <hip/hip_runtime.h>
#include <stdint.h>

#define M_DIM   8
#define N_DIM   8192
#define K_DIM   8192
#define GROUP   128
#define NGROUPS 64
#define KCHUNKS 8
#define GROUPS_PER_CHUNK 8
#define NTILES  512

typedef __attribute__((ext_vector_type(8))) int v8i;

__device__ __forceinline__ int koff_a(int j, int half) {
    return ((j & 1) << 2) + (((j >> 1) & 1) << 4) + ((j >> 2) << 5) + (half << 3);
}
__device__ __forceinline__ int koff_b(int j, int half) {
    return ((j & 1) << 2) + (((j >> 1) & 1) << 4) + ((j >> 2) << 5) + (half << 3);
}

__device__ __forceinline__ uint32_t pack4(const int4 w) {
    uint32_t t0 = __builtin_amdgcn_perm((uint32_t)w.y, (uint32_t)w.x, 0x00000400u);
    uint32_t t1 = __builtin_amdgcn_perm((uint32_t)w.w, (uint32_t)w.z, 0x00000400u);
    return __builtin_amdgcn_perm(t1, t0, 0x05040100u);
}

__device__ __forceinline__ v8i compute_group(const int* __restrict__ wrow,
                                             const uint32_t* __restrict__ swizA,
                                             int sg_base, int lane, int half) {
    v8i c = {0, 0, 0, 0, 0, 0, 0, 0};
    #pragma unroll
    for (int step = 0; step < 2; ++step) {
        const int sg    = sg_base + step;
        const int kbase = sg * 64;

        const v8i a = *(const v8i*)(swizA + ((size_t)sg * 32 + lane) * 8);

        v8i b;
        #pragma unroll
        for (int j = 0; j < 8; ++j) {
            const int4 w = *(const int4*)(wrow + kbase + koff_b(j, half));
            b[j] = (int)pack4(w);
        }
        c = __builtin_amdgcn_wmma_i32_16x16x64_iu8(true, a, true, b, c, false, false);
    }
    return c;
}

__global__ void qact_absmax_kernel(const float* __restrict__ x,
                                   float* __restrict__ sx) {
    __shared__ float red[256];
    const int m = blockIdx.x;
    const int t = threadIdx.x;
    float amax = 0.0f;
    for (int k = t; k < K_DIM; k += 256)
        amax = fmaxf(amax, fabsf(x[m * K_DIM + k]));
    red[t] = amax;
    __syncthreads();
    for (int off = 128; off > 0; off >>= 1) {
        if (t < off) red[t] = fmaxf(red[t], red[t + off]);
        __syncthreads();
    }
    if (t == 0) sx[m] = fmaxf(red[0] * (1.0f / 127.0f), 1e-8f);
}

__global__ void qact_pack_kernel(const float* __restrict__ x,
                                 const float* __restrict__ sx,
                                 uint32_t* __restrict__ swizA) {
    const int idx  = blockIdx.x * 256 + threadIdx.x;
    const int s    = idx >> 8;
    const int rem  = idx & 255;
    const int lane = rem >> 3;
    const int j    = rem & 7;
    const int m    = lane & 15;
    const int half = lane >> 4;
    const int k0   = s * 64 + koff_a(j, half);

    uint32_t packed = 0u;
    if (m < M_DIM) {
        const float inv_s = 1.0f / sx[m];
        #pragma unroll
        for (int b = 0; b < 4; ++b) {
            float q = rintf(x[m * K_DIM + k0 + b] * inv_s);
            q = fminf(fmaxf(q, -127.0f), 127.0f);
            packed |= ((uint32_t)((int)q & 0xFF)) << (8 * b);
        }
    }
    swizA[(s * 32 + lane) * 8 + j] = packed;
}

__global__ void __launch_bounds__(256)
wmma_main_kernel(const int* __restrict__ qw,
                 const float* __restrict__ scales,
                 const uint32_t* __restrict__ swizA,
                 const float* __restrict__ sx,
                 float* __restrict__ partials) {
    const int lane  = threadIdx.x & 31;
    const int wid   = threadIdx.x >> 5;
    const int gw    = blockIdx.x * 8 + wid;
    const int kc    = gw >> 9;
    const int ntile = gw & 511;
    const int nrow  = ntile * 16 + (lane & 15);
    const int half  = lane >> 4;

    float sxr[8];
    #pragma unroll
    for (int r = 0; r < 8; ++r) sxr[r] = sx[r];

    const int* __restrict__ wrow = qw + (size_t)nrow * K_DIM;
    const float* __restrict__ srow = scales + (size_t)nrow * NGROUPS
                                            + kc * GROUPS_PER_CHUNK;

    float facc[8] = {0.f, 0.f, 0.f, 0.f, 0.f, 0.f, 0.f, 0.f};

    v8i   c  = compute_group(wrow, swizA, kc * 16, lane, half);
    float sc = srow[0];
    for (int g = 1; g < GROUPS_PER_CHUNK; ++g) {
        const v8i   cn  = compute_group(wrow, swizA, kc * 16 + g * 2, lane, half);
        const float scn = srow[g];
        #pragma unroll
        for (int r = 0; r < 8; ++r)
            facc[r] += (float)c[r] * (sxr[r] * sc);
        c  = cn;
        sc = scn;
    }
    #pragma unroll
    for (int r = 0; r < 8; ++r)
        facc[r] += (float)c[r] * (sxr[r] * sc);

    if (lane < 16) {
        #pragma unroll
        for (int r = 0; r < 8; ++r)
            partials[((size_t)(kc * 8 + r)) * N_DIM + nrow] = facc[r];
    }
}

__global__ void out_reduce_kernel(const float* __restrict__ partials,
                                  const float* __restrict__ bias,
                                  float* __restrict__ out) {
    const int idx = blockIdx.x * 256 + threadIdx.x;
    const int m = idx >> 13;
    const int n = idx & (N_DIM - 1);
    float acc = bias[n];
    #pragma unroll
    for (int kc = 0; kc < KCHUNKS; ++kc)
        acc += partials[((size_t)(kc * 8 + m)) * N_DIM + n];
    out[idx] = acc;
}

extern "C" void kernel_launch(void* const* d_in, const int* in_sizes, int n_in,
                              void* d_out, int out_size, void* d_ws, size_t ws_size,
                              hipStream_t stream) {
    const float* x      = (const float*)d_in[0];
    const int*   qw     = (const int*)d_in[1];
    const float* scales = (const float*)d_in[2];
    const float* bias   = (const float*)d_in[3];
    float*       out    = (float*)d_out;

    char* ws = (char*)d_ws;
    float*    sx       = (float*)ws;
    uint32_t* swizA    = (uint32_t*)(ws + 256);
    float*    partials = (float*)(ws + 256 + 128 * 1024);

    qact_absmax_kernel<<<M_DIM, 256, 0, stream>>>(x, sx);
    qact_pack_kernel<<<128, 256, 0, stream>>>(x, sx, swizA);
    wmma_main_kernel<<<NTILES * KCHUNKS / 8, 256, 0, stream>>>(qw, scales, swizA,
                                                               sx, partials);
    out_reduce_kernel<<<(M_DIM * N_DIM) / 256, 256, 0, stream>>>(partials, bias, out);
}
